// QuantMatMul_31044023615647
// MI455X (gfx1250) — 8-bit matrix operand order repaired and checked by reading, compiled, not run on hardware
//
#include <hip/hip_runtime.h>
#include <stdint.h>

typedef __attribute__((ext_vector_type(8))) int v8i;

#define BH     64
#define LROWS  1024
#define KDIM   64
#define NCOLS  1024
#define QMAXF  255.0f

__device__ __forceinline__ uint32_t fenc(float f) {
  uint32_t u = __float_as_uint(f);
  return (u & 0x80000000u) ? ~u : (u | 0x80000000u);
}
__device__ __forceinline__ float fdec(uint32_t e) {
  uint32_t u = (e & 0x80000000u) ? (e ^ 0x80000000u) : ~e;
  return __uint_as_float(u);
}
__device__ __forceinline__ int bytesum(uint32_t w) {
  return (int)((w & 255u) + ((w >> 8) & 255u) + ((w >> 16) & 255u) + (w >> 24));
}

__global__ void init_minmax(uint32_t* mm) {
  int t = threadIdx.x;
  if (t < 4) mm[t] = (t & 1) ? 0u : 0xFFFFFFFFu;
}

__global__ __launch_bounds__(256) void minmax_kernel(const float4* __restrict__ x,
                                                     int n4, uint32_t* __restrict__ mm) {
  __shared__ uint32_t smn[256], smx[256];
  float lmin = 3.4e38f, lmax = -3.4e38f;
  for (int i = blockIdx.x * blockDim.x + threadIdx.x; i < n4; i += gridDim.x * blockDim.x) {
    float4 v = x[i];
    lmin = fminf(lmin, fminf(fminf(v.x, v.y), fminf(v.z, v.w)));
    lmax = fmaxf(lmax, fmaxf(fmaxf(v.x, v.y), fmaxf(v.z, v.w)));
  }
  int t = threadIdx.x;
  smn[t] = fenc(lmin);
  smx[t] = fenc(lmax);
  __syncthreads();
  for (int s = 128; s > 0; s >>= 1) {
    if (t < s) {
      uint32_t a = smn[t + s]; if (a < smn[t]) smn[t] = a;
      uint32_t b = smx[t + s]; if (b > smx[t]) smx[t] = b;
    }
    __syncthreads();
  }
  if (t == 0) { atomicMin(&mm[0], smn[0]); atomicMax(&mm[1], smx[0]); }
}

__global__ void finalize_params(const uint32_t* __restrict__ mm, float* __restrict__ params) {
  if (threadIdx.x == 0 && blockIdx.x == 0) {
    for (int t = 0; t < 2; ++t) {
      float mn = fdec(mm[2 * t]);
      float mx = fdec(mm[2 * t + 1]);
      float scale = fmaxf((mx - mn) / QMAXF, 1e-8f);
      float zp = rintf(-mn / scale);
      params[2 * t]     = scale;
      params[2 * t + 1] = zp;
    }
  }
}

__global__ __launch_bounds__(256) void quant_kernel(const float4* __restrict__ x,
                                                    uint32_t* __restrict__ q,
                                                    const float* __restrict__ params,
                                                    int pidx, int n4) {
  int i = blockIdx.x * blockDim.x + threadIdx.x;
  if (i >= n4) return;
  float scale = params[2 * pidx];
  float zp    = params[2 * pidx + 1];
  float4 v = x[i];
  float a0 = fminf(fmaxf(rintf(v.x / scale) + zp, 0.f), QMAXF);
  float a1 = fminf(fmaxf(rintf(v.y / scale) + zp, 0.f), QMAXF);
  float a2 = fminf(fmaxf(rintf(v.z / scale) + zp, 0.f), QMAXF);
  float a3 = fminf(fmaxf(rintf(v.w / scale) + zp, 0.f), QMAXF);
  q[i] = (uint32_t)a0 | ((uint32_t)a1 << 8) | ((uint32_t)a2 << 16) | ((uint32_t)a3 << 24);
}

__global__ __launch_bounds__(256) void rowsum_kernel(const uint32_t* __restrict__ q1d,
                                                     int* __restrict__ rs) {
  int row = blockIdx.x * blockDim.x + threadIdx.x;
  const uint32_t* p = q1d + (size_t)row * 16;
  int s = 0;
  #pragma unroll
  for (int j = 0; j < 16; ++j) s += bytesum(p[j]);
  rs[row] = s;
}

__global__ __launch_bounds__(256) void colsum_kernel(const uint32_t* __restrict__ q2d,
                                                     int* __restrict__ cs) {
  int gid = blockIdx.x * blockDim.x + threadIdx.x;
  int bh = gid >> 8;
  int cw = gid & 255;
  const uint32_t* p = q2d + (size_t)bh * (KDIM * NCOLS / 4) + cw;
  int s0 = 0, s1 = 0, s2 = 0, s3 = 0;
  #pragma unroll 8
  for (int k = 0; k < KDIM; ++k) {
    uint32_t w = p[(size_t)k * (NCOLS / 4)];
    s0 += (int)(w & 255u);
    s1 += (int)((w >> 8) & 255u);
    s2 += (int)((w >> 16) & 255u);
    s3 += (int)(w >> 24);
  }
  int* o = cs + (size_t)bh * NCOLS + cw * 4;
  o[0] = s0; o[1] = s1; o[2] = s2; o[3] = s3;
}

__global__ __launch_bounds__(256) void qmm_wmma_kernel(const uint8_t* __restrict__ q1,
                                                       const uint8_t* __restrict__ q2,
                                                       const int* __restrict__ rs,
                                                       const int* __restrict__ cs,
                                                       const float* __restrict__ params,
                                                       float* __restrict__ out) {
  __shared__ uint32_t raw32[2048];
  __shared__ uint32_t pack32[2048];

  const int tid = threadIdx.x;
  const int mb  = blockIdx.x;
  const int nb  = blockIdx.y;
  const int bh  = blockIdx.z;

  const int   zp1 = (int)params[1];
  const int   zp2 = (int)params[3];
  const float s12 = params[0] * params[2];

  {
    const uint8_t* bsrc = q2 + (size_t)bh * (KDIM * NCOLS) + (size_t)nb * 128;
    #pragma unroll
    for (int i = 0; i < 8; ++i) {
      int idx = tid + i * 256;
      int row = idx >> 5;
      int cw  = idx & 31;
      raw32[idx] = ((const uint32_t*)(bsrc + (size_t)row * NCOLS))[cw];
    }
  }
  __syncthreads();

  {
    const uint8_t* rb = (const uint8_t*)raw32;
    int n = tid >> 5, l = tid & 31;
    int m  = (n << 4) + (l & 15);
    int kb = (l & 16) ? 16 : 0;
    uint32_t w[8];
    #pragma unroll
    for (int j = 0; j < 4; ++j) {
      int k0 = kb + 4 * j;
      w[j] = (uint32_t)rb[(k0 + 0) * 128 + m]
           | ((uint32_t)rb[(k0 + 1) * 128 + m] << 8)
           | ((uint32_t)rb[(k0 + 2) * 128 + m] << 16)
           | ((uint32_t)rb[(k0 + 3) * 128 + m] << 24);
      int k1 = kb + 32 + 4 * j;
      w[4 + j] = (uint32_t)rb[(k1 + 0) * 128 + m]
               | ((uint32_t)rb[(k1 + 1) * 128 + m] << 8)
               | ((uint32_t)rb[(k1 + 2) * 128 + m] << 16)
               | ((uint32_t)rb[(k1 + 3) * 128 + m] << 24);
    }
    #pragma unroll
    for (int j = 0; j < 8; ++j) pack32[tid * 8 + j] = w[j];
  }
  __syncthreads();

  const int wave = tid >> 5;
  const int lane = tid & 31;

  v8i a;
  {
    const uint8_t* arow = q1 + (size_t)bh * (LROWS * KDIM)
                        + (size_t)(mb * 128 + wave * 16 + (lane & 15)) * KDIM
                        + ((lane & 16) ? 16 : 0);
    uint2 t0 = *(const uint2*)(arow + 0);
    uint2 t1 = *(const uint2*)(arow + 8);
    uint2 t2 = *(const uint2*)(arow + 32);
    uint2 t3 = *(const uint2*)(arow + 40);
    a[0] = (int)t0.x; a[1] = (int)t0.y;
    a[2] = (int)t1.x; a[3] = (int)t1.y;
    a[4] = (int)t2.x; a[5] = (int)t2.y;
    a[6] = (int)t3.x; a[7] = (int)t3.y;
  }

  const int rowbase = bh * LROWS + mb * 128 + wave * 16;
  const int half = (lane & 16) ? 8 : 0;
  const int kzz = KDIM * zp1 * zp2;
  int corrA[8];
  #pragma unroll
  for (int r = 0; r < 8; ++r)
    corrA[r] = kzz - zp2 * rs[rowbase + r + half];

  const int* csb = cs + bh * NCOLS + nb * 128 + (lane & 15);

  float* obase = out + (size_t)bh * (LROWS * NCOLS)
               + (size_t)(mb * 128 + wave * 16) * NCOLS
               + (size_t)nb * 128 + (lane & 15);

  #pragma unroll
  for (int n = 0; n < 8; ++n) {
    const int4* bp = (const int4*)((const uint8_t*)pack32 + (size_t)(n * 32 + lane) * 32);
    int4 lo = bp[0];
    int4 hi = bp[1];
    v8i b;
    b[0] = lo.x; b[1] = lo.y; b[2] = lo.z; b[3] = lo.w;
    b[4] = hi.x; b[5] = hi.y; b[6] = hi.z; b[7] = hi.w;

    v8i c = {0, 0, 0, 0, 0, 0, 0, 0};
    c = __builtin_amdgcn_wmma_i32_16x16x64_iu8(false, a, false, b, c, false, false);

    const int zc = zp1 * csb[n * 16];
    float* ocol = obase + n * 16;
    #pragma unroll
    for (int r = 0; r < 8; ++r) {
      int S = c[r] + corrA[r] - zc;
      __builtin_nontemporal_store(s12 * (float)S, ocol + (size_t)(r + half) * NCOLS);
    }
  }
}

extern "C" void kernel_launch(void* const* d_in, const int* in_sizes, int n_in,
                              void* d_out, int out_size, void* d_ws, size_t ws_size,
                              hipStream_t stream) {
  const float* x1 = (const float*)d_in[0];
  const float* x2 = (const float*)d_in[1];
  float* out = (float*)d_out;

  uint8_t*  ws     = (uint8_t*)d_ws;
  uint32_t* mm     = (uint32_t*)ws;
  float*    params = (float*)(ws + 16);
  uint8_t*  q1     = ws + 256;
  uint8_t*  q2     = q1 + (size_t)BH * LROWS * KDIM;
  int*      rs     = (int*)(q2 + (size_t)BH * KDIM * NCOLS);
  int*      cs     = rs + (size_t)BH * LROWS;

  const int n  = BH * LROWS * KDIM;
  const int n4 = n / 4;

  init_minmax<<<1, 4, 0, stream>>>(mm);
  minmax_kernel<<<256, 256, 0, stream>>>((const float4*)x1, n4, mm);
  minmax_kernel<<<256, 256, 0, stream>>>((const float4*)x2, n4, mm + 2);
  finalize_params<<<1, 1, 0, stream>>>(mm, params);
  quant_kernel<<<n4 / 256, 256, 0, stream>>>((const float4*)x1, (uint32_t*)q1, params, 0, n4);
  quant_kernel<<<n4 / 256, 256, 0, stream>>>((const float4*)x2, (uint32_t*)q2, params, 1, n4);
  rowsum_kernel<<<BH * LROWS / 256, 256, 0, stream>>>((const uint32_t*)q1, rs);
  colsum_kernel<<<BH * NCOLS / 4 / 256, 256, 0, stream>>>((const uint32_t*)q2, cs);

  dim3 grid(8, 8, BH);
  qmm_wmma_kernel<<<grid, 256, 0, stream>>>(q1, q2, rs, cs, params, out);
}
